// MultiHeadAttention_83640193123063
// MI455X (gfx1250) — matrix operand order repaired and checked by reading, not compiled, not run on hardware
//
#include <hip/hip_runtime.h>

// ---------------------------------------------------------------------------
// MI455X (gfx1250) multi-head attention forward.
// bf16 WMMA compute, TDM (tensor_load_to_lds) for all global->LDS staging.
// B=2, L=2048, D=2048, H=16, head_dim=128, rotary on first 64 channels/head.
// ---------------------------------------------------------------------------

typedef __bf16 bf16;
typedef __bf16 v16bf __attribute__((ext_vector_type(16)));
typedef float  v8f   __attribute__((ext_vector_type(8)));
typedef unsigned int su4 __attribute__((ext_vector_type(4)));
typedef unsigned int su8 __attribute__((ext_vector_type(8)));

union Frag { v16bf v; uint4 q[2]; };

#define BATCH 2
#define SEQ   2048
#define DM    2048
#define NH    16
#define HD    128

// ---------------------------------------------------------------------------
// Tensor Data Mover: 2D bf16 tile load, global -> LDS, with LDS row padding.
// Descriptor per CDNA5 ISA ch.8 (D# groups 0/1, 2D tensor => groups 2/3 NULL).
//   tile_w   : tile width in elements (contiguous dim)
//   tile_h   : tile height in rows
//   stride   : global row stride in elements
//   pad_int  : pad_interval code (3 => every 16 DW, 5 => every 64 DW)
//   pad_amt  : pad_amount  code (3 => 4 DWORDs = 16 bytes)
// Issued once per workgroup (wave 0); completion via s_wait_tensorcnt.
// ---------------------------------------------------------------------------
__device__ __forceinline__ void tdm_load_2d_bf16(const bf16* gptr,
                                                 unsigned lds_byte_off,
                                                 unsigned tile_w, unsigned tile_h,
                                                 unsigned stride,
                                                 unsigned pad_int, unsigned pad_amt) {
  unsigned long long ga = (unsigned long long)(size_t)gptr;
  su4 g0;
  g0[0] = 1u;                                   // count=1, user descriptor
  g0[1] = lds_byte_off;                         // lds_addr (bytes)
  g0[2] = (unsigned)(ga & 0xffffffffu);         // global_addr[31:0]
  g0[3] = (unsigned)((ga >> 32) & 0x01ffffffu)  // global_addr[56:32]
          | (2u << 30);                         // type = 2 ("image")
  su8 g1;
  g1[0] = (1u << 16)                            // data_size = 1 (2 bytes)
        | (1u << 20)                            // pad_enable
        | (pad_int << 22) | (pad_amt << 25);
  g1[1] = (stride & 0xffffu) << 16;             // tensor_dim0[15:0] (=stride)
  g1[2] = ((stride >> 16) & 0xffffu);           // tensor_dim0[31:16], dim1 lo16=0
  g1[3] = (1u << 14)                            // tensor_dim1[31:16] => dim1=2^30
        | ((tile_w & 0xffffu) << 16);           // tile_dim0
  g1[4] = (tile_h & 0xffffu);                   // tile_dim1 (tile_dim2 = 0)
  g1[5] = stride;                               // tensor_dim0_stride[31:0]
  g1[6] = 0u;                                   // stride[47:32]=0, dim1_stride lo=0
  g1[7] = 0u;
  asm volatile("tensor_load_to_lds %0, %1" :: "s"(g0), "s"(g1) : "memory");
}

// ---------------------------------------------------------------------------
// f32 -> bf16 conversion (grid-stride)
// ---------------------------------------------------------------------------
__global__ void f32_to_bf16_kernel(const float* __restrict__ in,
                                   bf16* __restrict__ out, int n) {
  int i = blockIdx.x * blockDim.x + threadIdx.x;
  int stride = gridDim.x * blockDim.x;
  for (; i < n; i += stride) out[i] = (bf16)in[i];
}

// ---------------------------------------------------------------------------
// f32 -> bf16 transposed copy of a DM x DM matrix: out[n][k] = in[k][n].
// 32x32 tiles through LDS so both sides stay coalesced.
// ---------------------------------------------------------------------------
__global__ __launch_bounds__(256)
void f32_to_bf16_transpose(const float* __restrict__ in, bf16* __restrict__ out) {
  __shared__ float t[32][33];
  int bx = blockIdx.x * 32, by = blockIdx.y * 32;
  int tx = threadIdx.x & 31, ty = threadIdx.x >> 5;  // 32 x 8
#pragma unroll
  for (int r = ty; r < 32; r += 8)
    t[r][tx] = in[(size_t)(by + r) * DM + bx + tx];
  __syncthreads();
#pragma unroll
  for (int r = ty; r < 32; r += 8)
    out[(size_t)(bx + r) * DM + by + tx] = (bf16)t[tx][r];
}

// ---------------------------------------------------------------------------
// Tiled bf16 GEMM:  Out(MxN) = A(MxK) * Bt(NxK)^T, f32 accumulate via WMMA.
// WG = 256 thr (8 waves), WG tile 128x64, wave tile 32x32 (2x2 WMMA tiles).
// A/Bt tiles staged in LDS by TDM with padded rows (stride 40 elems).
// mode 0: f32 row-major (M x N)
// mode 1: bf16 (B, H, L, HD) head-split, scaled
// mode 2: bf16 (B, H, HD, L) head-split transposed (for V)
// ---------------------------------------------------------------------------
__global__ __launch_bounds__(256)
void gemm_bf16_wmma(const bf16* __restrict__ A, const bf16* __restrict__ Bt,
                    void* __restrict__ Out, int M, int N, int Kd,
                    float scale, int mode) {
  constexpr int SAS = 40;            // 32 elems + 8 pad (TDM: 16DW + 4DW pad)
  constexpr int SBS = 40;
  extern __shared__ __align__(16) char smem[];
  bf16* sA  = (bf16*)smem;           // 128 x SAS   @ 0      (10240 B)
  bf16* sBt = (bf16*)(smem + 10240); //  64 x SBS   @ 10240  ( 5120 B)

  const int tid    = threadIdx.x;
  const int wave   = tid >> 5;
  const int lane   = tid & 31;
  const int lane16 = lane & 15;
  const int half   = lane >> 4;
  const int wm     = wave & 3;       // 4 wave rows
  const int wn     = wave >> 2;      // 2 wave cols
  const int gRow0  = blockIdx.x * 128;
  const int gCol0  = blockIdx.y * 64;

  v8f acc[2][2] = {};

  for (int kb = 0; kb < Kd; kb += 32) {
    __syncthreads();                 // previous tiles fully consumed
    if (wave == 0) {
      // A tile: 128 rows x 32 elems; Bt tile: 64 rows x 32 elems.
      tdm_load_2d_bf16(&A [(size_t)gRow0 * Kd + kb], 0u,     32u, 128u,
                       (unsigned)Kd, 3u, 3u);
      tdm_load_2d_bf16(&Bt[(size_t)gCol0 * Kd + kb], 10240u, 32u, 64u,
                       (unsigned)Kd, 3u, 3u);
      __builtin_amdgcn_s_wait_tensorcnt(0);
    }
    if (kb + 32 < Kd) {              // prefetch next K-slab into L2
      __builtin_prefetch(&A [(size_t)(gRow0 + (tid >> 1)) * Kd + kb + 32], 0, 1);
      __builtin_prefetch(&Bt[(size_t)(gCol0 + (tid >> 2)) * Kd + kb + 32], 0, 1);
    }
    __syncthreads();

    // A fragments: lane = row, K split {0..7,16..23} / {8..15,24..31} by half.
    Frag a[2], b[2];
#pragma unroll
    for (int ti = 0; ti < 2; ++ti) {
      int row = wm * 32 + ti * 16 + lane16;
      int k0 = half ? 8 : 0;
      a[ti].q[0] = *(const uint4*)&sA[row * SAS + k0];
      a[ti].q[1] = *(const uint4*)&sA[row * SAS + 16 + k0];
    }
    // B fragments: lane = col (N row of Bt), K split {0..7,16..23} / {8..15,24..31} by half.
#pragma unroll
    for (int tj = 0; tj < 2; ++tj) {
      int col = wn * 32 + tj * 16 + lane16;
      b[tj].q[0] = *(const uint4*)&sBt[col * SBS + half * 8];
      b[tj].q[1] = *(const uint4*)&sBt[col * SBS + half * 8 + 16];
    }
#pragma unroll
    for (int ti = 0; ti < 2; ++ti)
#pragma unroll
      for (int tj = 0; tj < 2; ++tj)
        acc[ti][tj] = __builtin_amdgcn_wmma_f32_16x16x32_bf16(
            false, a[ti].v, false, b[tj].v, (short)0, acc[ti][tj],
            false, false);
  }

  // C layout: VGPR r holds row r (+8 for upper lane half), col = lane16.
#pragma unroll
  for (int ti = 0; ti < 2; ++ti)
#pragma unroll
    for (int tj = 0; tj < 2; ++tj)
#pragma unroll
      for (int r = 0; r < 8; ++r) {
        int rowg = gRow0 + wm * 32 + ti * 16 + r + 8 * half;
        int colg = gCol0 + wn * 32 + tj * 16 + lane16;
        float val = acc[ti][tj][r] * scale;
        if (mode == 0) {
          ((float*)Out)[(size_t)rowg * N + colg] = val;
        } else {
          int bi = rowg / SEQ, l = rowg % SEQ;
          int h = colg >> 7, ch = colg & (HD - 1);
          if (mode == 1)
            ((bf16*)Out)[(((size_t)(bi * NH + h)) * SEQ + l) * HD + ch] =
                (bf16)val;
          else  // mode 2: transposed (B, H, HD, L) for V
            ((bf16*)Out)[(((size_t)(bi * NH + h)) * HD + ch) * SEQ + l] =
                (bf16)val;
        }
      }
}

// ---------------------------------------------------------------------------
// Rotary embedding on first 64 channels of each head of Q and K (in place).
// One thread per (b,h,l,pair).  Total threads = B*H*L*32 = 2^21.
// ---------------------------------------------------------------------------
__global__ void rope_kernel(bf16* __restrict__ Qh, bf16* __restrict__ Kh) {
  int idx = blockIdx.x * blockDim.x + threadIdx.x;
  int i  = idx & 31;                 // pair index, channels 2i, 2i+1
  int l  = (idx >> 5) & (SEQ - 1);   // sequence position
  int bh = idx >> 16;                // (b*H + h)
  float inv = __powf(10000.f, -(2.f * (float)i) / 64.f);
  float ang = (float)l * inv;
  float s, c;
  __sincosf(ang, &s, &c);
  size_t base = ((size_t)bh * SEQ + l) * HD + 2 * i;
  {
    float x1 = (float)Qh[base], x2 = (float)Qh[base + 1];
    Qh[base]     = (bf16)(x1 * c - x2 * s);
    Qh[base + 1] = (bf16)(x2 * c + x1 * s);
  }
  {
    float x1 = (float)Kh[base], x2 = (float)Kh[base + 1];
    Kh[base]     = (bf16)(x1 * c - x2 * s);
    Kh[base + 1] = (bf16)(x2 * c + x1 * s);
  }
}

// ---------------------------------------------------------------------------
// Flash attention (causal).  Grid: (L/128, B*H), 256 threads (8 waves).
// Wave w owns query rows [qblk*128 + w*16, +16).  Q fragments pinned in
// registers; per 32-key tile (TDM-staged K row-major + V channel-major):
// 8 WMMAs for S = Q*K^T, online softmax with cross-lane reductions,
// P repacked via per-wave LDS, 8 WMMAs for O += P*V.
// ---------------------------------------------------------------------------
__global__ __launch_bounds__(256)
void attn_wmma(const bf16* __restrict__ Qh, const bf16* __restrict__ Kh,
               const bf16* __restrict__ Vt, bf16* __restrict__ Yb) {
  constexpr int SKS = 136;  // 128 elems + 8 pad (TDM: 64DW + 4DW pad)
  constexpr int SVS = 40;   // 32 elems + 8 pad
  constexpr int SPS = 40;   // P staging row stride
  extern __shared__ __align__(16) char smem[];
  bf16* sK  = (bf16*)smem;            //  32 x SKS @ 0      (8704 B)
  bf16* sVt = (bf16*)(smem + 8704);   // 128 x SVS @ 8704   (10240 B)
  bf16* sP  = (bf16*)(smem + 18944);  // 8 waves x 16 x SPS (10240 B)

  const int tid    = threadIdx.x;
  const int wave   = tid >> 5;
  const int lane   = tid & 31;
  const int lane16 = lane & 15;
  const int half   = lane >> 4;
  const int bh = blockIdx.y;
  const int b  = bh >> 4;
  const int h  = bh & 15;
  const int rowbase = blockIdx.x * 128 + wave * 16;

  const bf16* Q  = Qh + (size_t)bh * SEQ * HD;
  const bf16* Kp = Kh + (size_t)bh * SEQ * HD;
  const bf16* Vp = Vt + (size_t)bh * HD * SEQ;   // (HD, SEQ) channel-major
  bf16* sPw = sP + wave * 16 * SPS;

  // Q fragments for all 4 K-chunks (16x32 each), pinned in registers.
  Frag qf[4];
  {
    int row = rowbase + lane16;
    int k0 = half ? 8 : 0;
#pragma unroll
    for (int ch = 0; ch < 4; ++ch) {
      qf[ch].q[0] = *(const uint4*)&Q[(size_t)row * HD + ch * 32 + k0];
      qf[ch].q[1] = *(const uint4*)&Q[(size_t)row * HD + ch * 32 + 16 + k0];
    }
  }

  v8f o[8] = {};
  float mrun[8], lrun[8];
#pragma unroll
  for (int r = 0; r < 8; ++r) { mrun[r] = -1e30f; lrun[r] = 0.f; }

  const int jmax = (blockIdx.x * 128 + 127) >> 5;  // inclusive, causal extent
  for (int jt = 0; jt <= jmax; ++jt) {
    __syncthreads();
    if (wave == 0) {
      // K tile: 32 keys x 128 ch, row-major.  V tile: 128 ch x 32 keys.
      tdm_load_2d_bf16(&Kp[(size_t)(jt * 32) * HD], 0u,    128u, 32u,
                       (unsigned)HD, 5u, 3u);
      tdm_load_2d_bf16(&Vp[jt * 32],                8704u,  32u, 128u,
                       (unsigned)SEQ, 3u, 3u);
      __builtin_amdgcn_s_wait_tensorcnt(0);
    }
    __syncthreads();

    if (jt * 32 <= rowbase + 15) {  // wave-uniform causal skip
      // ---- S = Q * K^T  (two 16x16 tiles covering 32 keys) ----
      v8f c0 = {}, c1 = {};
#pragma unroll
      for (int ch = 0; ch < 4; ++ch) {
        Frag bk0, bk1;
        int kr = half ? 8 : 0;
        bk0.q[0] = *(const uint4*)&sK[(lane16)*SKS + ch * 32 + kr];
        bk0.q[1] = *(const uint4*)&sK[(lane16)*SKS + ch * 32 + kr + 16];
        bk1.q[0] = *(const uint4*)&sK[(16 + lane16) * SKS + ch * 32 + kr];
        bk1.q[1] = *(const uint4*)&sK[(16 + lane16) * SKS + ch * 32 + kr + 16];
        c0 = __builtin_amdgcn_wmma_f32_16x16x32_bf16(
            false, qf[ch].v, false, bk0.v, (short)0, c0, false, false);
        c1 = __builtin_amdgcn_wmma_f32_16x16x32_bf16(
            false, qf[ch].v, false, bk1.v, (short)0, c1, false, false);
      }

      // ---- causal mask + online softmax (row stats per VGPR slot) ----
#pragma unroll
      for (int r = 0; r < 8; ++r) {
        int qrow = rowbase + r + 8 * half;
        int key0 = jt * 32 + lane16;
        float s0 = (key0 <= qrow)      ? c0[r] : -1e8f;
        float s1 = (key0 + 16 <= qrow) ? c1[r] : -1e8f;
        float t = fmaxf(s0, s1);
#pragma unroll
        for (int off = 1; off < 16; off <<= 1)
          t = fmaxf(t, __shfl_xor(t, off, 32));   // stays within 16-lane half
        float mnew  = fmaxf(mrun[r], t);
        float alpha = __expf(mrun[r] - mnew);
        float p0 = __expf(s0 - mnew);
        float p1 = __expf(s1 - mnew);
        float rs = p0 + p1;
#pragma unroll
        for (int off = 1; off < 16; off <<= 1)
          rs += __shfl_xor(rs, off, 32);
        lrun[r] = lrun[r] * alpha + rs;
        mrun[r] = mnew;
#pragma unroll
        for (int ot = 0; ot < 8; ++ot) o[ot][r] *= alpha;
        int prow = r + 8 * half;  // repack P through per-wave LDS staging
        sPw[prow * SPS + lane16]      = (bf16)p0;
        sPw[prow * SPS + 16 + lane16] = (bf16)p1;
      }

      // ---- O += P * V  (P as A-fragment, V tile as 8 B-fragments) ----
      Frag pf;
      {
        int k0 = half ? 8 : 0;
        pf.q[0] = *(const uint4*)&sPw[lane16 * SPS + k0];
        pf.q[1] = *(const uint4*)&sPw[lane16 * SPS + 16 + k0];
      }
#pragma unroll
      for (int ot = 0; ot < 8; ++ot) {
        Frag vf;
        int n = ot * 16 + lane16;  // channel row in sVt, keys contiguous
        vf.q[0] = *(const uint4*)&sVt[n * SVS + half * 8];
        vf.q[1] = *(const uint4*)&sVt[n * SVS + half * 8 + 16];
        o[ot] = __builtin_amdgcn_wmma_f32_16x16x32_bf16(
            false, pf.v, false, vf.v, (short)0, o[ot], false, false);
      }
    }
  }

  // ---- normalize and write to (B, L, D) bf16 for the wo GEMM ----
#pragma unroll
  for (int ot = 0; ot < 8; ++ot)
#pragma unroll
    for (int r = 0; r < 8; ++r) {
      int qrow = rowbase + r + 8 * half;
      int ch = ot * 16 + lane16;
      float val = o[ot][r] / lrun[r];
      Yb[((size_t)(b * SEQ + qrow)) * DM + h * HD + ch] = (bf16)val;
    }
}

// ---------------------------------------------------------------------------
// Host-side orchestration
// ---------------------------------------------------------------------------
extern "C" void kernel_launch(void* const* d_in, const int* in_sizes, int n_in,
                              void* d_out, int out_size, void* d_ws,
                              size_t ws_size, hipStream_t stream) {
  (void)in_sizes; (void)n_in; (void)out_size; (void)ws_size;
  const float* x  = (const float*)d_in[0];
  // d_in[1] is the causal mask; tril by construction, applied analytically.
  const float* wq = (const float*)d_in[2];
  const float* wk = (const float*)d_in[3];
  const float* wv = (const float*)d_in[4];
  const float* wo = (const float*)d_in[5];

  const size_t NX = (size_t)BATCH * SEQ * DM;  // 8,388,608
  const size_t NW = (size_t)DM * DM;           // 4,194,304

  bf16* xb  = (bf16*)d_ws;     // x, bf16
  bf16* wqt = xb + NX;         // wq^T, bf16 (N-major)
  bf16* wkt = wqt + NW;
  bf16* wvt = wkt + NW;
  bf16* wot = wvt + NW;
  bf16* qh  = wot + NW;        // (B,H,L,HD)
  bf16* kh  = qh + NX;         // (B,H,L,HD)
  bf16* vt  = kh + NX;         // (B,H,HD,L)
  bf16* yb  = vt + NX;         // (B,L,D)

  dim3 blk(256);
  f32_to_bf16_kernel<<<4096, blk, 0, stream>>>(x, xb, (int)NX);
  dim3 tgrid(DM / 32, DM / 32);  // (64, 64)
  f32_to_bf16_transpose<<<tgrid, blk, 0, stream>>>(wq, wqt);
  f32_to_bf16_transpose<<<tgrid, blk, 0, stream>>>(wk, wkt);
  f32_to_bf16_transpose<<<tgrid, blk, 0, stream>>>(wv, wvt);
  f32_to_bf16_transpose<<<tgrid, blk, 0, stream>>>(wo, wot);

  dim3 ggrid(BATCH * SEQ / 128, DM / 64);  // (32, 32)
  const unsigned gemm_lds = 15360;
  const float qscale = 0.08838834764831845f;  // 1/sqrt(128)
  gemm_bf16_wmma<<<ggrid, blk, gemm_lds, stream>>>(xb, wqt, qh,
      BATCH * SEQ, DM, DM, qscale, 1);
  gemm_bf16_wmma<<<ggrid, blk, gemm_lds, stream>>>(xb, wkt, kh,
      BATCH * SEQ, DM, DM, 1.0f, 1);
  gemm_bf16_wmma<<<ggrid, blk, gemm_lds, stream>>>(xb, wvt, vt,
      BATCH * SEQ, DM, DM, 1.0f, 2);

  rope_kernel<<<8192, blk, 0, stream>>>(qh, kh);  // B*H*L*32 = 2^21 threads

  dim3 agrid(SEQ / 128, BATCH * NH);  // (16, 32)
  attn_wmma<<<agrid, blk, 29184, stream>>>(qh, kh, vt, yb);

  gemm_bf16_wmma<<<ggrid, blk, gemm_lds, stream>>>(yb, wot, d_out,
      BATCH * SEQ, DM, DM, 1.0f, 0);
}
